// SetGNN_50302656971597
// MI455X (gfx1250) — matrix operand order repaired and checked by reading, not compiled, not run on hardware
//
#include <hip/hip_runtime.h>
#include <hip/hip_bf16.h>

// ---------------- constants ----------------
#define EPSF     1e-5f
#define NNODES   100000
#define NHE      20000
#define NINC     640000
#define NYMAX    20000

// ---------------- WMMA types ----------------
typedef __attribute__((ext_vector_type(16))) __bf16 bf16x16;
typedef __attribute__((ext_vector_type(8)))  float  f32x8;

union FragAB { bf16x16 v; unsigned int u[8]; };

__device__ __forceinline__ unsigned short bf16bits(float f) {
  union { float f; unsigned int u; } x; x.f = f;
  unsigned int u = x.u;
  u += 0x7fffu + ((u >> 16) & 1u);          // round-to-nearest-even
  return (unsigned short)(u >> 16);
}

// A-fragment: 16x32 bf16 tile, row-major source with row stride `stride`.
// ISA 7.12.2 layout: lanes 0-15 -> M, half=lane>>4 selects K sub-block.
__device__ __forceinline__ void load_fragA(FragAB& a, const unsigned short* base,
                                           int stride, int kb, int lane) {
  const int m = lane & 15, half = lane >> 4;
#pragma unroll
  for (int j = 0; j < 8; ++j) {
    const int ka = (j < 4) ? (half * 8 + 2 * j) : (16 + half * 8 + 2 * (j - 4));
    a.u[j] = *(const unsigned int*)(base + m * stride + kb + ka);
  }
}

// B-fragment: 32x16 bf16; weights stored transposed [N][K] so K is contiguous.
__device__ __forceinline__ void load_fragB(FragAB& b, const unsigned short* wt,
                                           int K, int n0, int kb, int lane) {
  const int n = n0 + (lane & 15), half = lane >> 4;
#pragma unroll
  for (int j = 0; j < 8; ++j) {
    const int kk = kb + ((j < 4) ? (half * 8 + 2 * j) : (16 + half * 8 + 2 * (j - 4)));
    b.u[j] = *(const unsigned int*)(wt + n * K + kk);
  }
}

template <int K>
__device__ __forceinline__ void gemm16(const unsigned short* Ash,
                                       const unsigned short* Wt,
                                       int n0, int lane, f32x8& c) {
#pragma unroll
  for (int ks = 0; ks < K / 32; ++ks) {
    FragAB a, b;
    load_fragA(a, Ash, K, ks * 32, lane);
    load_fragB(b, Wt, K, n0, ks * 32, lane);
    c = __builtin_amdgcn_wmma_f32_16x16x32_bf16(false, a.v, false, b.v,
                                                (short)0, c, false, false);
  }
}

// ---------------- fused AllSet MLP: ln0 -> W1 -> relu -> ln1 -> W2 -> relu ----------------
__global__ __launch_bounds__(256) void mlp128_kernel(
    const float* __restrict__ in, float* __restrict__ out, int nrows,
    const unsigned short* __restrict__ w1t, const unsigned short* __restrict__ w2t,
    const float* __restrict__ ln0g, const float* __restrict__ ln0b,
    const float* __restrict__ b1,
    const float* __restrict__ ln1g, const float* __restrict__ ln1b,
    const float* __restrict__ b2)
{
  __shared__ float          fbuf[16][128];
  __shared__ unsigned short abf[16][128];
  __shared__ float          rmu[16], rrs[16];

  const int tid  = threadIdx.x;
  const int lane = tid & 31;
  const int wv   = tid >> 5;                // 0..7 -> 16-col tile
  const int half = lane >> 4;
  const int row0 = blockIdx.x * 16;
  const int n    = wv * 16 + (lane & 15);

  // stage 0: load raw 16x128 tile
#pragma unroll
  for (int e = 0; e < 8; ++e) {
    int idx = tid + 256 * e, r = idx >> 7, c = idx & 127;
    int gr = row0 + r;
    fbuf[r][c] = (gr < nrows) ? in[(long)gr * 128 + c] : 0.f;
  }
  __syncthreads();

  // stage 1: ln0 stats
  if (tid < 16) {
    float m = 0.f;
    for (int c = 0; c < 128; ++c) m += fbuf[tid][c];
    m *= (1.f / 128.f);
    float v = 0.f;
    for (int c = 0; c < 128; ++c) { float d = fbuf[tid][c] - m; v += d * d; }
    rmu[tid] = m;
    rrs[tid] = rsqrtf(v * (1.f / 128.f) + EPSF);
  }
  __syncthreads();

  // stage 2: normalize -> bf16 A operand
#pragma unroll
  for (int e = 0; e < 8; ++e) {
    int idx = tid + 256 * e, r = idx >> 7, c = idx & 127;
    float v = (fbuf[r][c] - rmu[r]) * rrs[r] * ln0g[c] + ln0b[c];
    abf[r][c] = bf16bits(v);
  }
  __syncthreads();

  // stage 3: layer 1 WMMA, +b1, relu -> fbuf
  {
    f32x8 c = {0.f, 0.f, 0.f, 0.f, 0.f, 0.f, 0.f, 0.f};
    gemm16<128>(&abf[0][0], w1t, wv * 16, lane, c);
    const float bias = b1[n];
#pragma unroll
    for (int r = 0; r < 8; ++r)
      fbuf[r + half * 8][n] = fmaxf(c[r] + bias, 0.f);
  }
  __syncthreads();

  // stage 4: ln1 stats
  if (tid < 16) {
    float m = 0.f;
    for (int c = 0; c < 128; ++c) m += fbuf[tid][c];
    m *= (1.f / 128.f);
    float v = 0.f;
    for (int c = 0; c < 128; ++c) { float d = fbuf[tid][c] - m; v += d * d; }
    rmu[tid] = m;
    rrs[tid] = rsqrtf(v * (1.f / 128.f) + EPSF);
  }
  __syncthreads();

  // stage 5: normalize -> bf16 A operand (layer 2)
#pragma unroll
  for (int e = 0; e < 8; ++e) {
    int idx = tid + 256 * e, r = idx >> 7, c = idx & 127;
    float v = (fbuf[r][c] - rmu[r]) * rrs[r] * ln1g[c] + ln1b[c];
    abf[r][c] = bf16bits(v);
  }
  __syncthreads();

  // stage 6: layer 2 WMMA, +b2, relu (outer relu of HalfNLHconv), store
  {
    f32x8 c = {0.f, 0.f, 0.f, 0.f, 0.f, 0.f, 0.f, 0.f};
    gemm16<128>(&abf[0][0], w2t, wv * 16, lane, c);
    const float bias = b2[n];
#pragma unroll
    for (int r = 0; r < 8; ++r) {
      int gr = row0 + r + half * 8;
      if (gr < nrows) out[(long)gr * 128 + n] = fmaxf(c[r] + bias, 0.f);
    }
  }
}

// ---------------- classifier: concat(256) -> W1 -> relu -> ln1 -> W2(128x2) ----------------
__global__ __launch_bounds__(256) void clf_kernel(
    const float* __restrict__ xa, const float* __restrict__ xb,
    const int* __restrict__ nyp,
    const unsigned short* __restrict__ w1t,   // [128][256] bf16 (N-major)
    const float* __restrict__ b1,
    const float* __restrict__ ln1g, const float* __restrict__ ln1b,
    const float* __restrict__ w2,             // [128][2] f32
    const float* __restrict__ b2,
    float* __restrict__ out)
{
  __shared__ unsigned short abf[16][256];
  __shared__ float          hbuf[16][128];
  __shared__ float          rmu[16], rrs[16];

  const int tid  = threadIdx.x;
  const int lane = tid & 31;
  const int wv   = tid >> 5;
  const int half = lane >> 4;
  const int row0 = blockIdx.x * 16;
  const int n    = wv * 16 + (lane & 15);

#pragma unroll
  for (int e = 0; e < 16; ++e) {
    int idx = tid + 256 * e, r = idx >> 8, c = idx & 255;
    int gr = row0 + r;
    float v = 0.f;
    if (gr < NYMAX)
      v = (c < 128) ? xa[(long)gr * 128 + c] : xb[(long)gr * 128 + (c - 128)];
    abf[r][c] = bf16bits(v);
  }
  __syncthreads();

  {
    f32x8 c = {0.f, 0.f, 0.f, 0.f, 0.f, 0.f, 0.f, 0.f};
    gemm16<256>(&abf[0][0], w1t, wv * 16, lane, c);
    const float bias = b1[n];
#pragma unroll
    for (int r = 0; r < 8; ++r)
      hbuf[r + half * 8][n] = fmaxf(c[r] + bias, 0.f);
  }
  __syncthreads();

  if (tid < 16) {
    float m = 0.f;
    for (int c = 0; c < 128; ++c) m += hbuf[tid][c];
    m *= (1.f / 128.f);
    float v = 0.f;
    for (int c = 0; c < 128; ++c) { float d = hbuf[tid][c] - m; v += d * d; }
    rmu[tid] = m;
    rrs[tid] = rsqrtf(v * (1.f / 128.f) + EPSF);
  }
  __syncthreads();

#pragma unroll
  for (int e = 0; e < 8; ++e) {
    int idx = tid + 256 * e, r = idx >> 7, c = idx & 127;
    hbuf[r][c] = (hbuf[r][c] - rmu[r]) * rrs[r] * ln1g[c] + ln1b[c];
  }
  __syncthreads();

  if (tid < 32) {
    int r = tid >> 1, col = tid & 1;
    float s = 0.f;
    for (int k = 0; k < 128; ++k) s += hbuf[r][k] * w2[k * 2 + col];
    int gr = row0 + r;
    int ny = *nyp; if (ny > NYMAX) ny = NYMAX;
    if (gr < ny) out[(long)gr * 2 + col] = s + b2[col];
  }
}

// ---------------- small helper kernels ----------------
__global__ void zero_kernel(float* __restrict__ p, long n) {
  long i = (long)blockIdx.x * blockDim.x + threadIdx.x;
  long st = (long)gridDim.x * blockDim.x;
  for (; i < n; i += st) p[i] = 0.f;
}

__global__ void init_int_kernel(int* p, int v) { *p = v; }

__global__ void imin_kernel(const int* __restrict__ a, int n, int* __restrict__ out) {
  int i = blockIdx.x * blockDim.x + threadIdx.x;
  int st = gridDim.x * blockDim.x;
  int m = 0x7fffffff;
  for (; i < n; i += st) m = min(m, a[i]);
  atomicMin(out, m);
}

// W[k][n] f32 -> WT[n][k] bf16
__global__ void cvtT_kernel(const float* __restrict__ w, unsigned short* __restrict__ wt,
                            int K, int N) {
  int total = K * N;
  for (int i = blockIdx.x * blockDim.x + threadIdx.x; i < total;
       i += gridDim.x * blockDim.x) {
    int k = i / N, n = i - k * N;
    wt[n * K + k] = bf16bits(w[i]);
  }
}

__global__ void count_kernel(const int* __restrict__ dst, float* __restrict__ cnt,
                             const int* __restrict__ minp, int subMin, int n) {
  int i = blockIdx.x * blockDim.x + threadIdx.x;
  if (i < n) {
    int d = dst[i] - (subMin ? *minp : 0);
    atomicAdd(&cnt[d], 1.0f);
  }
}

__global__ __launch_bounds__(256) void scatter_kernel(
    const float* __restrict__ xe, const int* __restrict__ src,
    const int* __restrict__ dst, const float* __restrict__ nrm,
    float* __restrict__ s, const int* __restrict__ minp,
    int subSrc, int subDst, int n_inc)
{
  long t = (long)blockIdx.x * 256 + threadIdx.x;
  int i = (int)(t >> 5);
  if (i >= n_inc) return;
  int cq = ((int)t & 31) << 2;
  int mn = *minp;
  int si = src[i] - (subSrc ? mn : 0);
  int di = dst[i] - (subDst ? mn : 0);
  float nv = nrm[i];
  const float4 v = *(const float4*)(xe + (long)si * 128 + cq);
  float* sp = s + (long)di * 128 + cq;
  atomicAdd(sp + 0, nv * v.x);
  atomicAdd(sp + 1, nv * v.y);
  atomicAdd(sp + 2, nv * v.z);
  atomicAdd(sp + 3, nv * v.w);
}

__global__ void divide_kernel(float* __restrict__ s, const float* __restrict__ cnt, long total) {
  long i = (long)blockIdx.x * blockDim.x + threadIdx.x;
  long st = (long)gridDim.x * blockDim.x;
  for (; i < total; i += st) s[i] = s[i] / fmaxf(cnt[i >> 7], 1.0f);
}

// gnorm: pass 1 col-sums + raw sumsq
__global__ __launch_bounds__(128) void gnorm_stats(const float* __restrict__ x, int n,
                                                   float* __restrict__ g) {
  const int c = threadIdx.x;  // 128
  float cs = 0.f, sq = 0.f;
  for (int r = blockIdx.x; r < n; r += gridDim.x) {
    float v = x[(long)r * 128 + c];
    cs += v; sq += v * v;
  }
  atomicAdd(&g[c], cs);
  __shared__ float sh[128];
  sh[c] = sq; __syncthreads();
  for (int s = 64; s > 0; s >>= 1) { if (c < s) sh[c] += sh[c + s]; __syncthreads(); }
  if (c == 0) atomicAdd(&g[128], sh[0]);
}

// gnorm: finalize mean[c] -> g[130+c], scale -> g[129]
__global__ __launch_bounds__(128) void gnorm_final(float* __restrict__ g, float inv_n, float n_f) {
  const int c = threadIdx.x;
  float mean = g[c] * inv_n;
  __shared__ float sh[128];
  sh[c] = mean * mean; __syncthreads();
  for (int s = 64; s > 0; s >>= 1) { if (c < s) sh[c] += sh[c + s]; __syncthreads(); }
  if (c == 0) {
    float cent = g[128] - n_f * sh[0];
    g[129] = rsqrtf(EPSF + fmaxf(cent * inv_n, 0.f));
  }
  g[130 + c] = mean;
}

// gnorm apply (+ optional eval-mode BN + relu into H)
__global__ void gnorm_apply(float* __restrict__ x, float* __restrict__ h,
                            const float* __restrict__ g,
                            const float* __restrict__ bg, const float* __restrict__ bb,
                            long total) {
  const float bnsc = rsqrtf(1.0f + EPSF);
  long i = (long)blockIdx.x * blockDim.x + threadIdx.x;
  long st = (long)gridDim.x * blockDim.x;
  for (; i < total; i += st) {
    int c = (int)(i & 127);
    float v = (x[i] - g[130 + c]) * g[129];
    x[i] = v;
    if (h) h[i] = fmaxf(v * bnsc * bg[c] + bb[c], 0.f);
  }
}

// ---------------- host orchestration ----------------
extern "C" void kernel_launch(void* const* d_in, const int* in_sizes, int n_in,
                              void* d_out, int out_size, void* d_ws, size_t ws_size,
                              hipStream_t stream) {
  (void)in_sizes; (void)n_in; (void)out_size; (void)ws_size;

  const float* x_in     = (const float*)d_in[0];
  const int*   e_nodes  = (const int*)d_in[1];
  const int*   e_hedges = (const int*)d_in[2];
  const float* nrm      = (const float*)d_in[3];
  const int*   n_y_p    = (const int*)d_in[4];
  auto P = [&](int i) { return (const float*)d_in[i]; };
  // params flat order (insertion order): mlp = {ln0g,ln0b,W1,b1,ln1g,ln1b,W2,b2}
  // v2e0.enc@5, v2e0.dec@13, e2v0.enc@21, e2v0.dec@29, v2e1.enc@37, v2e1.dec@45,
  // bnV2E0@53/54, bnE2V0@55/56, clf@57..64
  const int mbase[6] = {5, 13, 21, 29, 37, 45};

  // workspace layout
  unsigned short* wbf = (unsigned short*)d_ws;             // 12*16384 + 32768 bf16
  size_t wbytes = ((size_t)(12 * 16384 + 32768) * 2 + 255) & ~(size_t)255;
  float* fbase   = (float*)((char*)d_ws + wbytes);
  float* buf_enc = fbase;                                  // 100000*128
  float* sbuf    = buf_enc + (long)NNODES * 128;           // 100000*128
  float* cnt     = sbuf + (long)NNODES * 128;              // 100000
  float* x1buf   = cnt + NNODES;                           // 20000*128
  float* h1buf   = x1buf + (long)NHE * 128;                // 20000*128
  float* h2buf   = h1buf + (long)NHE * 128;                // 100000*128
  float* gstat   = h2buf + (long)NNODES * 128;             // 512
  int*   ehmin   = (int*)(gstat + 512);

  float* out_score = (float*)d_out;                        // [20000,2]
  float* out_edge  = out_score + (long)NYMAX * 2;          // [20000,128]
  float* out_node  = out_edge + (long)NHE * 128;           // [100000,128]

  // --- hedge index rebase: min(edge_hedges) ---
  init_int_kernel<<<1, 1, 0, stream>>>(ehmin, 0x7fffffff);
  imin_kernel<<<2500, 256, 0, stream>>>(e_hedges, NINC, ehmin);

  // --- weights -> bf16, transposed to [N][K] ---
  for (int m = 0; m < 6; ++m) {
    cvtT_kernel<<<64, 256, 0, stream>>>(P(mbase[m] + 2), wbf + (2 * m + 0) * 16384, 128, 128);
    cvtT_kernel<<<64, 256, 0, stream>>>(P(mbase[m] + 6), wbf + (2 * m + 1) * 16384, 128, 128);
  }
  cvtT_kernel<<<128, 256, 0, stream>>>(P(59), wbf + 12 * 16384, 256, 128);
  auto W = [&](int slot) { return (const unsigned short*)(wbf + slot * 16384); };

  auto run_mlp = [&](const float* in, float* out, int nrows, int bi,
                     const unsigned short* w1t, const unsigned short* w2t) {
    mlp128_kernel<<<(nrows + 15) / 16, 256, 0, stream>>>(
        in, out, nrows, w1t, w2t,
        P(bi + 0), P(bi + 1), P(bi + 3), P(bi + 4), P(bi + 5), P(bi + 7));
  };
  auto zero = [&](float* p, long n) {
    int blk = (int)((n + 255) / 256); if (blk > 4096) blk = 4096;
    zero_kernel<<<blk, 256, 0, stream>>>(p, n);
  };
  auto halfagg = [&](const float* xe, const int* src, const int* dst,
                     int subSrc, int subDst, int nseg) {
    zero(sbuf, (long)nseg * 128);
    zero(cnt, nseg);
    count_kernel<<<(NINC + 255) / 256, 256, 0, stream>>>(dst, cnt, ehmin, subDst, NINC);
    scatter_kernel<<<(int)(((long)NINC * 32 + 255) / 256), 256, 0, stream>>>(
        xe, src, dst, nrm, sbuf, ehmin, subSrc, subDst, NINC);
    long tot = (long)nseg * 128;
    divide_kernel<<<(int)((tot + 255) / 256) > 4096 ? 4096 : (int)((tot + 255) / 256),
                    256, 0, stream>>>(sbuf, cnt, tot);
  };
  auto gnorm = [&](float* X, float* H, int n, const float* bg, const float* bb) {
    zero(gstat, 512);
    gnorm_stats<<<2048, 128, 0, stream>>>(X, n, gstat);
    gnorm_final<<<1, 128, 0, stream>>>(gstat, 1.0f / (float)n, (float)n);
    long tot = (long)n * 128;
    int blk = (int)((tot + 255) / 256); if (blk > 8192) blk = 8192;
    gnorm_apply<<<blk, 256, 0, stream>>>(X, H, gstat, bg, bb, tot);
  };

  // ---- V2E layer 0 ----
  run_mlp(x_in, buf_enc, NNODES, 5, W(0), W(1));               // enc
  halfagg(buf_enc, e_nodes, e_hedges, 0, 1, NHE);              // scatter-mean nodes->hedges
  run_mlp(sbuf, x1buf, NHE, 13, W(2), W(3));                   // dec
  gnorm(x1buf, h1buf, NHE, P(53), P(54));                      // x1 (=vec0), h1=relu(bn(x1))

  // ---- E2V layer 0 (reversed) ----
  run_mlp(h1buf, buf_enc, NHE, 21, W(4), W(5));
  halfagg(buf_enc, e_hedges, e_nodes, 1, 0, NNODES);           // hedges->nodes
  run_mlp(sbuf, out_node, NNODES, 29, W(6), W(7));
  gnorm(out_node, h2buf, NNODES, P(55), P(56));                // node_feat, h2

  // ---- V2E layer 1 ----
  run_mlp(h2buf, buf_enc, NNODES, 37, W(8), W(9));
  halfagg(buf_enc, e_nodes, e_hedges, 0, 1, NHE);
  run_mlp(sbuf, out_edge, NHE, 45, W(10), W(11));
  gnorm(out_edge, nullptr, NHE, nullptr, nullptr);             // edge_feat (no BN after)

  // ---- classifier on concat([vec0, x3]) ----
  clf_kernel<<<(NYMAX + 15) / 16, 256, 0, stream>>>(
      x1buf, out_edge, n_y_p, (const unsigned short*)(wbf + 12 * 16384),
      P(60), P(61), P(62), P(63), P(64), out_score);
}
